// HypeTKGEncoder_51823075393725
// MI455X (gfx1250) — matrix operand order repaired and checked by reading, not compiled, not run on hardware
//
#include <hip/hip_runtime.h>
#include <hip/hip_bf16.h>
#include <math.h>
#include <stdint.h>

#define NUM_ENT 50000
#define RAW_ENT 45000
#define NREL2   1000
#define D       200
#define NE      750000
#define NS      100000
#define BSZ     4096
#define MAXQ    6
#define TMAXV   365

#define KR      224            // padded region stride (7 k-blocks of 32)
#define NPAD    208            // padded N = 13 * 16
#define NT      13             // N tiles per GEMM (fixed for this workload)
#define MPADE   50016          // NUM_ENT padded to 16
#define MPADR   1008           // NREL2 padded to 16
#define KIN     (3*KR)         // 672 : [(pre+teagg)/deg | spre/sdeg | x]
#define KPROJ   (2*KR)         // 448 : [x[sub] | te[time]]
#define PBREG   (NT*7*32*8)    // u32 elems per packed-B 224-K region

typedef __attribute__((ext_vector_type(16))) __bf16 v16bf;
typedef __attribute__((ext_vector_type(8)))  float  v8f;

// ---------------------------------------------------------------- time table
__global__ void k_build_te(float* te, const float* w, const float* ph) {
  int i = blockIdx.x * blockDim.x + threadIdx.x;
  if (i >= TMAXV * D) return;
  int t = i / D, d = i % D;
  te[i] = cosf((float)t * w[d] + ph[d]);
}

// ------------------------------------------------- x-independent edge passes
__global__ void k_edge_te_deg(const int* dst, const int* et, const float* te,
                              float* teagg, float* deg, int n) {
  int lane = threadIdx.x & 31;
  int e = blockIdx.x * (blockDim.x >> 5) + (threadIdx.x >> 5);
  if (e >= n) return;
  int dv = dst[e], t = et[e];
  const float* trow = te + (size_t)t * D;
  float* arow = teagg + (size_t)dv * D;
  for (int i = lane; i < D; i += 32) atomicAdd(&arow[i], trow[i]);
  if (lane == 0) atomicAdd(&deg[dv], 1.0f);
}

__global__ void k_count(const int* dst, float* deg, int n) {
  int i = blockIdx.x * blockDim.x + threadIdx.x;
  if (i < n) atomicAdd(&deg[dst[i]], 1.0f);
}

// -------------------------------------------------- per-layer edge scatter
// one wave per edge: pre[dst] += x[src] * r[etype]
__global__ void k_edge_msg(const int* src, const int* dst, const int* et,
                           const float* x, const float* r, float* pre, int n) {
  int lane = threadIdx.x & 31;
  int e = blockIdx.x * (blockDim.x >> 5) + (threadIdx.x >> 5);
  if (e >= n) return;
  int sv = src[e], dv = dst[e], tv = et[e];
  const float* xs = x + (size_t)sv * D;
  const float* rt = r + (size_t)tv * D;
  float* pd = pre + (size_t)dv * D;
  for (int i = lane; i < D; i += 32) atomicAdd(&pd[i], xs[i] * rt[i]);
}

// ------------------------------------------- build fused GEMM A matrix (bf16)
__global__ void k_build_in(__bf16* in, const float* pre, const float* teagg,
                           const float* deg, const float* spre, const float* sdeg,
                           const float* x) {
  long i = (long)blockIdx.x * blockDim.x + threadIdx.x;
  if (i >= (long)MPADE * KIN) return;
  int v = (int)(i / KIN), k = (int)(i % KIN);
  int g = k / KR, c = k % KR;
  float val = 0.0f;
  if (v < NUM_ENT && c < D) {
    size_t o = (size_t)v * D + c;
    if (g == 0)      val = (pre[o] + teagg[o]) / fmaxf(deg[v], 1.0f);
    else if (g == 1) val = spre[o] / fmaxf(sdeg[v], 1.0f);
    else             val = x[o];
  }
  in[i] = (__bf16)val;
}

// generic f32[srcR][srcC] -> zero-padded row-major bf16[dstR][dstC] (A packs)
__global__ void k_pack(__bf16* dst, const float* src, int srcR, int srcC,
                       int dstR, int dstC) {
  long i = (long)blockIdx.x * blockDim.x + threadIdx.x;
  if (i >= (long)dstR * dstC) return;
  int r = (int)(i / dstC), c = (int)(i % dstC);
  float v = (r < srcR && c < srcC) ? src[(size_t)r * srcC + c] : 0.0f;
  dst[i] = (__bf16)v;
}

// pack one 224-K region of a weight into the WMMA B-fragment layout:
// dst[((nt*KBtot + kbOff + kbl)*32 + lane)*8 + j] = {B[kk][n], B[kk+1][n]}
// with kk = kbl*32 + (j>>2)*16 + (lane>>4)*8 + 2(j&3), n = nt*16 + (lane&15)
__global__ void k_pack_bfrag(uint32_t* dst, const float* src, int srcR, int srcC,
                             int kbOff, int KBtot) {
  long i = (long)blockIdx.x * blockDim.x + threadIdx.x;
  if (i >= (long)PBREG) return;
  int j    = (int)(i & 7);
  int lane = (int)((i >> 3) & 31);
  int kbl  = (int)((i >> 8) % 7);
  int nt   = (int)(i / (7 * 256));
  int half = lane >> 4, col = lane & 15;
  int n  = nt * 16 + col;
  int kk = kbl * 32 + (j >> 2) * 16 + half * 8 + 2 * (j & 3);
  float b0 = (kk     < srcR && n < srcC) ? src[(size_t)kk * srcC + n] : 0.0f;
  float b1 = (kk + 1 < srcR && n < srcC) ? src[(size_t)(kk + 1) * srcC + n] : 0.0f;
  union { __bf16 h[2]; uint32_t u; } p;
  p.h[0] = (__bf16)b0;  p.h[1] = (__bf16)b1;
  dst[(((size_t)nt * KBtot + kbOff + kbl) * 32 + lane) * 8 + j] = p.u;
}

// projA[BSZ][KPROJ]: region0 = x2[sub[b]], region1 = TE[time[b]]
__global__ void k_proj_a(__bf16* pa, const float* x, const float* te,
                         const int* sub, const int* tm) {
  long i = (long)blockIdx.x * blockDim.x + threadIdx.x;
  if (i >= (long)BSZ * KPROJ) return;
  int b = (int)(i / KPROJ), k = (int)(i % KPROJ);
  int g = k / KR, c = k % KR;
  float v = 0.0f;
  if (c < D)
    v = (g == 0) ? x[(size_t)sub[b] * D + c] : te[(size_t)tm[b] * D + c];
  pa[i] = (__bf16)v;
}

// ------------------------------------------------------- WMMA bf16 GEMM
// Each wave computes a 16-row x 208-col strip: A fragment loaded once per
// k-block (two aligned b128s), 13 resident packed-B fragments -> 13 WMMAs.
// KB is a template constant so all B-fragment addresses are one base pointer
// plus immediate offsets; __launch_bounds__(256,1) lets the 13 f32 accumulator
// fragments (104 VGPRs) live in registers instead of scratch.
template<int KB>
__global__ __launch_bounds__(256, 1)
void k_gemm_wmma(const __bf16* __restrict__ A,
                 const uint32_t* __restrict__ Bp,
                 float* __restrict__ C,
                 int aRows, int M, int N,
                 int ldc, const float* bias, int act) {
  int lane = threadIdx.x & 31;
  int wave = threadIdx.x >> 5;
  int mBase = (blockIdx.x * (blockDim.x >> 5) + wave) * 16;
  if (mBase >= aRows) return;            // uniform per wave: EXEC stays full
  int half = lane >> 4;
  int col  = lane & 15;
  const uint32_t* ar = (const uint32_t*)(A + (size_t)(mBase + col) * (KB * 32))
                       + (half << 2);
  const uint32_t* bbase = Bp + (size_t)lane * 8;

  v8f acc[NT];
#pragma unroll
  for (int t = 0; t < NT; ++t)
    acc[t] = (v8f){0.f, 0.f, 0.f, 0.f, 0.f, 0.f, 0.f, 0.f};

#pragma unroll 1
  for (int kb = 0; kb < KB; ++kb) {
    // A fragment: u32-pair runs [kb*16+half*4 .. +3] and [+8 .. +11]
    union { uint4 q[2]; v16bf v; } ua;
    const uint4* a4 = (const uint4*)(ar + (kb << 4));
    ua.q[0] = a4[0];
    ua.q[1] = a4[2];
    const uint32_t* brow = bbase + kb * 256;
#pragma unroll
    for (int t = 0; t < NT; ++t) {
      union { uint4 q[2]; v16bf v; } ub;
      const uint4* b4 = (const uint4*)(brow + t * (KB * 256));  // imm offsets
      ub.q[0] = b4[0];
      ub.q[1] = b4[1];
      acc[t] = __builtin_amdgcn_wmma_f32_16x16x32_bf16(
          false, ua.v, false, ub.v, (short)0, acc[t], false, false);
    }
  }

#pragma unroll
  for (int t = 0; t < NT; ++t) {
    int n = t * 16 + col;
    if (n < N) {
      float bv = bias ? bias[n] : 0.0f;
#pragma unroll
      for (int v = 0; v < 8; ++v) {
        int m = mBase + (half << 3) + v;   // C layout: M = v + 8*half
        if (m < M) {
          float cv = acc[t][v] + bv;
          if (act) cv = tanhf(cv);
          C[(size_t)m * ldc + n] = cv;
        }
      }
    }
  }
}

// ---------------------------------------------------------------- gathers
__global__ void k_gather_rows(float* out, const float* tbl, const int* idx, int rows) {
  long i = (long)blockIdx.x * blockDim.x + threadIdx.x;
  if (i >= (long)rows * D) return;
  int r = (int)(i / D), d = (int)(i % D);
  out[i] = tbl[(size_t)idx[r] * D + d];
}

__global__ void k_qual(float* out, const float* r2, const float* x2, const int* quals) {
  long i = (long)blockIdx.x * blockDim.x + threadIdx.x;
  if (i >= (long)BSZ * 2 * MAXQ * D) return;
  int b = (int)(i / (2 * MAXQ * D));
  int j = (int)((i / D) % (2 * MAXQ));
  int d = (int)(i % D);
  int id = quals[b * 2 * MAXQ + j];
  out[i] = (j & 1) ? x2[(size_t)id * D + d] : r2[(size_t)id * D + d];
}

__global__ void k_copy(float* out, const float* in, long n) {
  long i = (long)blockIdx.x * blockDim.x + threadIdx.x;
  if (i < n) out[i] = in[i];
}

// =========================================================================
extern "C" void kernel_launch(void* const* d_in, const int* in_sizes, int n_in,
                              void* d_out, int out_size, void* d_ws, size_t ws_size,
                              hipStream_t stream) {
  const int*   sub         = (const int*)  d_in[0];
  const int*   rel         = (const int*)  d_in[1];
  const int*   timev       = (const int*)  d_in[2];
  const int*   quals       = (const int*)  d_in[3];
  const int*   edge_index  = (const int*)  d_in[4];   // [2][E]
  const int*   edge_type   = (const int*)  d_in[5];
  const int*   edge_time   = (const int*)  d_in[6];
  const int*   static_index= (const int*)  d_in[7];   // [2][S]
  const int*   static_type = (const int*)  d_in[8];
  const float* init_embed  = (const float*)d_in[9];
  const float* init_rel    = (const float*)d_in[10];
  const float* Wm[2]  = {(const float*)d_in[11], (const float*)d_in[15]};
  const float* Ws[2]  = {(const float*)d_in[12], (const float*)d_in[16]};
  const float* Wr[2]  = {(const float*)d_in[13], (const float*)d_in[17]};
  const float* Wst[2] = {(const float*)d_in[14], (const float*)d_in[18]};
  const float* basis  = (const float*)d_in[19];
  const float* phase  = (const float*)d_in[20];
  const float* proj_W = (const float*)d_in[21];
  const float* proj_b = (const float*)d_in[22];
  float* out = (float*)d_out;

  const int* e_src = edge_index;          const int* e_dst = edge_index + NE;
  const int* s_src = static_index;        const int* s_dst = static_index + NS;

  // ---- workspace carve-out -------------------------------------------------
  size_t off = 0;
  char* base = (char*)d_ws;
  auto alloc = [&](size_t bytes) -> void* {
    off = (off + 255) & ~(size_t)255;
    void* p = base + off;  off += bytes;  return p;
  };
  float*    TE    = (float*)   alloc((size_t)TMAXV * D * 4);
  float*    teagg = (float*)   alloc((size_t)NUM_ENT * D * 4);
  float*    deg   = (float*)   alloc((size_t)NUM_ENT * 4);
  float*    sdeg  = (float*)   alloc((size_t)NUM_ENT * 4);
  float*    pre   = (float*)   alloc((size_t)NUM_ENT * D * 4);
  float*    spre  = (float*)   alloc((size_t)NUM_ENT * D * 4);
  float*    x1    = (float*)   alloc((size_t)NUM_ENT * D * 4);
  float*    x2    = (float*)   alloc((size_t)NUM_ENT * D * 4);
  float*    r1    = (float*)   alloc((size_t)NREL2 * D * 4);
  float*    r2    = (float*)   alloc((size_t)NREL2 * D * 4);
  __bf16*   IN    = (__bf16*)  alloc((size_t)MPADE * KIN * 2);
  uint32_t* Wc    = (uint32_t*)alloc((size_t)NT * 21 * 256 * 4);  // KB=21
  __bf16*   rA    = (__bf16*)  alloc((size_t)MPADR * KR * 2);
  uint32_t* WrC   = (uint32_t*)alloc((size_t)NT * 7 * 256 * 4);   // KB=7
  __bf16*   pA    = (__bf16*)  alloc((size_t)BSZ * KPROJ * 2);
  uint32_t* pB    = (uint32_t*)alloc((size_t)NT * 14 * 256 * 4);  // KB=14
  (void)ws_size;

  const int T = 256;
  auto nb = [](long n, int t) { return (int)((n + t - 1) / t); };
  const int EB = NE / 8;                  // wave-per-edge blocks (8 waves/blk)
  const int SB = nb(NS, 8);
  const int PBB = nb(PBREG, T);

  // ---- x-independent precompute -------------------------------------------
  hipMemsetAsync(teagg, 0, (size_t)NUM_ENT * D * 4, stream);
  hipMemsetAsync(deg,   0, (size_t)NUM_ENT * 4, stream);
  hipMemsetAsync(sdeg,  0, (size_t)NUM_ENT * 4, stream);
  hipMemsetAsync(pre,   0, (size_t)NUM_ENT * D * 4, stream);
  hipMemsetAsync(spre,  0, (size_t)NUM_ENT * D * 4, stream);

  k_build_te<<<nb((long)TMAXV * D, T), T, 0, stream>>>(TE, basis, phase);
  k_edge_te_deg<<<EB, T, 0, stream>>>(e_dst, edge_time, TE, teagg, deg, NE);
  k_count<<<nb(NS, T), T, 0, stream>>>(s_dst, sdeg, NS);

  // ---- two conv layers -----------------------------------------------------
  const float* xin = init_embed;  const float* rin = init_rel;
  float* xouts[2] = {x1, x2};     float* routs[2] = {r1, r2};
  dim3 gMain((MPADE / 16 + 7) / 8);
  dim3 gRel ((MPADR / 16 + 7) / 8);

  for (int l = 0; l < 2; ++l) {
    if (l == 1) {
      hipMemsetAsync(pre,  0, (size_t)NUM_ENT * D * 4, stream);
      hipMemsetAsync(spre, 0, (size_t)NUM_ENT * D * 4, stream);
    }
    k_edge_msg<<<EB, T, 0, stream>>>(e_src, e_dst, edge_type, xin, rin, pre, NE);
    k_edge_msg<<<SB, T, 0, stream>>>(s_src, s_dst, static_type, xin, rin, spre, NS);
    // fused A = [(pre+teagg)/deg | spre/sdeg | x], B = frag-packed [Wm;Wst;Ws]
    k_build_in<<<nb((long)MPADE * KIN, T), T, 0, stream>>>(IN, pre, teagg, deg,
                                                           spre, sdeg, xin);
    k_pack_bfrag<<<PBB, T, 0, stream>>>(Wc, Wm[l],  D, D, 0,  21);
    k_pack_bfrag<<<PBB, T, 0, stream>>>(Wc, Wst[l], D, D, 7,  21);
    k_pack_bfrag<<<PBB, T, 0, stream>>>(Wc, Ws[l],  D, D, 14, 21);
    k_gemm_wmma<21><<<gMain, T, 0, stream>>>(IN, Wc, xouts[l], MPADE, NUM_ENT, D,
                                             D, nullptr, /*tanh*/1);
    // r_out = r @ Wr
    k_pack<<<nb((long)MPADR * KR, T), T, 0, stream>>>(rA, rin, NREL2, D, MPADR, KR);
    k_pack_bfrag<<<PBB, T, 0, stream>>>(WrC, Wr[l], D, D, 0, 7);
    k_gemm_wmma<7><<<gRel, T, 0, stream>>>(rA, WrC, routs[l], MPADR, NREL2, D,
                                           D, nullptr, 0);
    xin = xouts[l];  rin = routs[l];
  }

  // ---- outputs -------------------------------------------------------------
  const size_t O_SUB  = 0;
  const size_t O_REL  = O_SUB  + (size_t)BSZ * D;
  const size_t O_QUAL = O_REL  + (size_t)BSZ * D;
  const size_t O_XRAW = O_QUAL + (size_t)BSZ * 2 * MAXQ * D;
  const size_t O_TIME = O_XRAW + (size_t)RAW_ENT * D;

  // sub_emb = [x2[sub] | TE[time]] @ proj_W + proj_b
  k_proj_a<<<nb((long)BSZ * KPROJ, T), T, 0, stream>>>(pA, x2, TE, sub, timev);
  k_pack_bfrag<<<PBB, T, 0, stream>>>(pB, proj_W,         D, D, 0, 14);
  k_pack_bfrag<<<PBB, T, 0, stream>>>(pB, proj_W + D * D, D, D, 7, 14);
  dim3 gProj((BSZ / 16 + 7) / 8);
  k_gemm_wmma<14><<<gProj, T, 0, stream>>>(pA, pB, out + O_SUB, BSZ, BSZ, D,
                                           D, proj_b, 0);

  k_gather_rows<<<nb((long)BSZ * D, T), T, 0, stream>>>(out + O_REL,  r2, rel,   BSZ);
  k_qual<<<nb((long)BSZ * 2 * MAXQ * D, T), T, 0, stream>>>(out + O_QUAL, r2, x2, quals);
  k_copy<<<nb((long)RAW_ENT * D, T), T, 0, stream>>>(out + O_XRAW, x2, (long)RAW_ENT * D);
  k_gather_rows<<<nb((long)BSZ * D, T), T, 0, stream>>>(out + O_TIME, TE, timev, BSZ);
}
